// InvoiceGNN_28836410425606
// MI455X (gfx1250) — compile-verified, matrix operand order checked by reading, not run on hardware
//
#include <hip/hip_runtime.h>
#include <hip/hip_bf16.h>

typedef __attribute__((ext_vector_type(16))) _Float16 v16h;
typedef __attribute__((ext_vector_type(8)))  _Float16 v8h;
typedef __attribute__((ext_vector_type(8)))  float    v8f;

#define N_NODES 50000
#define N_EDGES 800000
#define D_IN    128
#define D_HID   256

// ---------------------------------------------------------------------------
// Degree: cnt[dst[e]] += 1
// ---------------------------------------------------------------------------
__global__ void k_degree(const int* __restrict__ dst, float* __restrict__ cnt, int E) {
    int e = blockIdx.x * blockDim.x + threadIdx.x;
    if (e < E) atomicAdd(&cnt[dst[e]], 1.0f);
}

// ---------------------------------------------------------------------------
// Scatter-sum: one wave32 per edge, float4 per lane per 128 features.
// Feature matrices fit in the 192MB L2, so gathers + atomics resolve in L2.
// ---------------------------------------------------------------------------
template<int D>
__global__ void k_scatter(const int* __restrict__ src, const int* __restrict__ dst,
                          const float* __restrict__ feat, float* __restrict__ msg, int E) {
    int wave = (blockIdx.x * blockDim.x + threadIdx.x) >> 5;
    int lane = threadIdx.x & 31;
    if (wave >= E) return;
    int s = src[wave];
    int d = dst[wave];
    const float4* fs = (const float4*)(feat + (size_t)s * D);
    float* md = msg + (size_t)d * D;
#pragma unroll
    for (int j = 0; j < D / 128; ++j) {
        float4 v = fs[lane + 32 * j];
        int base = (lane + 32 * j) * 4;
        atomicAdd(&md[base + 0], v.x);
        atomicAdd(&md[base + 1], v.y);
        atomicAdd(&md[base + 2], v.z);
        atomicAdd(&md[base + 3], v.w);
    }
}

// ---------------------------------------------------------------------------
// agg_h = f16( msg / max(cnt,1) )  -- f16 only; GEMM consumes f16 directly
// ---------------------------------------------------------------------------
__global__ void k_div_h(const float* __restrict__ msg, const float* __restrict__ cnt,
                        _Float16* __restrict__ agg_h, int n, int shift) {
    int i = blockIdx.x * blockDim.x + threadIdx.x;
    if (i < n) {
        float c = cnt[i >> shift];
        c = c > 1.0f ? c : 1.0f;
        agg_h[i] = (_Float16)(msg[i] / c);
    }
}

// ---------------------------------------------------------------------------
// Elementwise f32 -> f16 copy (for x)
// ---------------------------------------------------------------------------
__global__ void k_f32_to_f16(const float* __restrict__ in, _Float16* __restrict__ out, int n) {
    int i = blockIdx.x * blockDim.x + threadIdx.x;
    if (i < n) out[i] = (_Float16)in[i];
}

// ---------------------------------------------------------------------------
// Pack W [Kh x N] (row-major f32) into fragment-ordered f16:
//   Bp[((tn*chunks + c)*32 + lane)*16 + i] = f16( W[k, tn*16 + (lane&15)] )
//   k = c*32 + (v>>2)*16 + (lane>>4)*8 + (v&3)*2 + (i&1),  v = i>>1
// so a wave reads a whole 32x16 B fragment with two contiguous b128 loads.
// ---------------------------------------------------------------------------
__global__ void k_pack_w(const float* __restrict__ W, _Float16* __restrict__ Bp,
                         int Kh, int N) {
    int chunks = Kh >> 5;
    int total = (N >> 4) * chunks * 32 * 16;
    int idx = blockIdx.x * blockDim.x + threadIdx.x;
    if (idx >= total) return;
    int i    = idx & 15;
    int lane = (idx >> 4) & 31;
    int cc   = (idx >> 9) % chunks;
    int tn   = (idx >> 9) / chunks;
    int v    = i >> 1;
    int khalf = (lane >> 4) << 3;
    int k   = (cc << 5) + ((v >> 2) << 4) + khalf + ((v & 3) << 1) + (i & 1);
    int col = (tn << 4) + (lane & 15);
    Bp[idx] = (_Float16)W[(size_t)k * N + col];
}

// ---------------------------------------------------------------------------
// Fused SAGE GEMM: out = act( A0 @ W0 + A1 @ W1 + bias )
//   A0h/A1h: row-major f16 [M x Kh]; B0p/B1p: fragment-packed f16.
// One wave per 16x16 D-tile. Inner loop: 4 x b128 loads + 1 v_wmma.
// Optionally also writes an f16 row-major copy (feeds next layer's A).
// ---------------------------------------------------------------------------
__global__ void k_sage_gemm(const _Float16* __restrict__ A0h, const _Float16* __restrict__ A1h,
                            const _Float16* __restrict__ B0p, const _Float16* __restrict__ B1p,
                            const float* __restrict__ bias, float* __restrict__ out,
                            _Float16* __restrict__ out_h,
                            int M, int Kh, int N, int do_relu) {
    int lane = threadIdx.x & 31;
    int wave = threadIdx.x >> 5;
    int tilesN = N >> 4;
    int numTiles = (M >> 4) * tilesN;
    int tile = blockIdx.x * (blockDim.x >> 5) + wave;
    if (tile >= numTiles) return;
    int tm = tile / tilesN;
    int tn = tile - tm * tilesN;

    int mrow  = (tm << 4) + (lane & 15);   // A row owned by this lane
    int ncol  = (tn << 4) + (lane & 15);   // B/D column owned by this lane
    int khalf = (lane >> 4) << 3;          // 0 or 8

    v8f acc = {0.f, 0.f, 0.f, 0.f, 0.f, 0.f, 0.f, 0.f};

    int chunks = Kh >> 5;
    for (int s = 0; s < 2; ++s) {
        const _Float16* Arow = (s ? A1h : A0h) + (size_t)mrow * Kh + khalf;
        const _Float16* Bl   = (s ? B1p : B0p) + ((size_t)tn * chunks * 32 + lane) * 16;
        for (int c = 0; c < chunks; ++c) {
            v8h alo = *(const v8h*)(Arow + (c << 5));        // k = kbase+khalf .. +7
            v8h ahi = *(const v8h*)(Arow + (c << 5) + 16);   // k = kbase+16+khalf .. +7
            v8h blo = *(const v8h*)(Bl + (size_t)c * 32 * 16);
            v8h bhi = *(const v8h*)(Bl + (size_t)c * 32 * 16 + 8);
            v16h af, bf;
#pragma unroll
            for (int j = 0; j < 8; ++j) {
                af[j] = alo[j]; af[j + 8] = ahi[j];
                bf[j] = blo[j]; bf[j + 8] = bhi[j];
            }
            acc = __builtin_amdgcn_wmma_f32_16x16x32_f16(
                false, af, false, bf, (short)0, acc, false, false);
        }
    }

    float bv = bias[ncol];
    int rowbase = (tm << 4) + khalf;       // C/D layout: VGPR r -> row r + (lane>>4)*8
#pragma unroll
    for (int r = 0; r < 8; ++r) {
        float v = acc[r] + bv;
        if (do_relu) v = v > 0.0f ? v : 0.0f;
        size_t off = (size_t)(rowbase + r) * N + ncol;
        out[off] = v;
        if (out_h) out_h[off] = (_Float16)v;
    }
}

// ---------------------------------------------------------------------------
// Final projection: out[i] = h2[i,:] . Wo + bo   (wave32 per node)
// ---------------------------------------------------------------------------
__global__ void k_out(const float* __restrict__ h, const float* __restrict__ Wo,
                      const float* __restrict__ bo, float* __restrict__ out,
                      int M, int K) {
    int node = (blockIdx.x * blockDim.x + threadIdx.x) >> 5;
    int lane = threadIdx.x & 31;
    if (node >= M) return;
    const float* row = h + (size_t)node * K;
    float s = 0.f;
    for (int j = lane; j < K; j += 32) s += row[j] * Wo[j];
#pragma unroll
    for (int off = 16; off; off >>= 1) s += __shfl_xor(s, off, 32);
    if (lane == 0) out[node] = s + bo[0];
}

// ---------------------------------------------------------------------------
extern "C" void kernel_launch(void* const* d_in, const int* in_sizes, int n_in,
                              void* d_out, int out_size, void* d_ws, size_t ws_size,
                              hipStream_t stream) {
    const float* x   = (const float*)d_in[0];
    const int*   ei  = (const int*)d_in[1];   // int32 [2, E], row-major
    const float* W1l = (const float*)d_in[2];
    const float* W1r = (const float*)d_in[3];
    const float* b1  = (const float*)d_in[4];
    const float* W2l = (const float*)d_in[5];
    const float* W2r = (const float*)d_in[6];
    const float* b2  = (const float*)d_in[7];
    const float* Wo  = (const float*)d_in[8];
    const float* bo  = (const float*)d_in[9];

    const int Nn = N_NODES, E = N_EDGES;
    const int* src = ei;
    const int* dst = ei + E;

    // ---- workspace layout ----
    // f32: msg[Nn*256] | cnt[Nn] | h1[Nn*256] | h2[Nn*256]
    float* msg = (float*)d_ws;
    float* cnt = msg + (size_t)Nn * D_HID;
    float* h1  = cnt + Nn;
    float* h2  = h1 + (size_t)Nn * D_HID;
    // f16: x_h[Nn*128] | agg_h[Nn*256] | h1_h[Nn*256] | packed weights
    _Float16* x_h   = (_Float16*)(h2 + (size_t)Nn * D_HID);
    _Float16* agg_h = x_h + (size_t)Nn * D_IN;
    _Float16* h1_h  = agg_h + (size_t)Nn * D_HID;
    _Float16* Bp1l  = h1_h + (size_t)Nn * D_HID;
    _Float16* Bp1r  = Bp1l + (size_t)D_IN * D_HID;
    _Float16* Bp2l  = Bp1r + (size_t)D_IN * D_HID;
    _Float16* Bp2r  = Bp2l + (size_t)D_HID * D_HID;

    const int TPB = 256;

    // ---- one-time-per-launch prep: degree, x->f16, weight packing ----
    hipMemsetAsync(cnt, 0, (size_t)Nn * sizeof(float), stream);
    k_degree<<<(E + TPB - 1) / TPB, TPB, 0, stream>>>(dst, cnt, E);
    k_f32_to_f16<<<((Nn * D_IN) + TPB - 1) / TPB, TPB, 0, stream>>>(x, x_h, Nn * D_IN);
    k_pack_w<<<((D_IN  * D_HID) + TPB - 1) / TPB, TPB, 0, stream>>>(W1l, Bp1l, D_IN,  D_HID);
    k_pack_w<<<((D_IN  * D_HID) + TPB - 1) / TPB, TPB, 0, stream>>>(W1r, Bp1r, D_IN,  D_HID);
    k_pack_w<<<((D_HID * D_HID) + TPB - 1) / TPB, TPB, 0, stream>>>(W2l, Bp2l, D_HID, D_HID);
    k_pack_w<<<((D_HID * D_HID) + TPB - 1) / TPB, TPB, 0, stream>>>(W2r, Bp2r, D_HID, D_HID);

    const int numTiles = (Nn / 16) * (D_HID / 16);   // 3125 * 16 = 50000
    const int gemmBlocks = (numTiles + 7) / 8;       // 8 waves / block

    // ---- layer 1 ----
    hipMemsetAsync(msg, 0, (size_t)Nn * D_IN * sizeof(float), stream);
    k_scatter<D_IN><<<(E * 32 + TPB - 1) / TPB, TPB, 0, stream>>>(src, dst, x, msg, E);
    k_div_h<<<((Nn * D_IN) + TPB - 1) / TPB, TPB, 0, stream>>>(msg, cnt, agg_h, Nn * D_IN, 7);
    k_sage_gemm<<<gemmBlocks, TPB, 0, stream>>>(agg_h, x_h, Bp1l, Bp1r, b1,
                                                h1, h1_h, Nn, D_IN, D_HID, 1);

    // ---- layer 2 ----
    hipMemsetAsync(msg, 0, (size_t)Nn * D_HID * sizeof(float), stream);
    k_scatter<D_HID><<<(E * 32 + TPB - 1) / TPB, TPB, 0, stream>>>(src, dst, h1, msg, E);
    k_div_h<<<((Nn * D_HID) + TPB - 1) / TPB, TPB, 0, stream>>>(msg, cnt, agg_h, Nn * D_HID, 8);
    k_sage_gemm<<<gemmBlocks, TPB, 0, stream>>>(agg_h, h1_h, Bp2l, Bp2r, b2,
                                                h2, (_Float16*)nullptr, Nn, D_HID, D_HID, 1);

    // ---- output projection ----
    k_out<<<(Nn * 32 + TPB - 1) / TPB, TPB, 0, stream>>>(h2, Wo, bo, (float*)d_out,
                                                         Nn, D_HID);
}
